// EdgeEmbedding_40570261078234
// MI455X (gfx1250) — matrix operand order repaired and checked by reading, not compiled, not run on hardware
//
#include <hip/hip_runtime.h>

typedef _Float16 v16h __attribute__((ext_vector_type(16)));
typedef float    v8f  __attribute__((ext_vector_type(8)));

#define EDGES_PER_WAVE 16
#define WAVES_PER_BLOCK 8
#define PI_F 3.14159265358979f
#define CUTOFF_F 5.0f

// ---------------------------------------------------------------------------
// Prologue: repack Wz (128x64 f32) and Wr (32x192 f32) into f16 WMMA B-tiles.
// Tile layout in ws: tile*512 halfs, lane*16 halfs each (one v16h per lane).
// B (32x16, f16) register layout per CDNA5 ISA: lanes 0-15 hold K=0..7,16..23,
// lanes 16-31 hold K=8..15,24..31; element j within a lane = K offset j + (j & 8) + 8*h.
// Tiles 0..15  : Wz, tile = kc*4 + t  (kc = K-chunk of 32, t = 16-col group)
// Tiles 16..27 : Wr, tile-16 = u      (u = 16-col group of the 192 cols)
// ---------------------------------------------------------------------------
__global__ void pack_weights_kernel(const float* __restrict__ Wz,
                                    const float* __restrict__ Wr,
                                    _Float16* __restrict__ bw) {
    int tile = blockIdx.x;
    int lane = threadIdx.x;            // 0..31
    int h    = lane >> 4;              // lane half
    int col  = lane & 15;              // B column within 16-wide tile
    _Float16* dst = bw + (size_t)tile * 512 + (size_t)lane * 16;
    if (tile < 16) {
        int kc = tile >> 2, t = tile & 3;
        #pragma unroll
        for (int j = 0; j < 16; ++j) {
            int k = 32 * kc + j + (j & 8) + 8 * h;        // B k-mapping
            dst[j] = (_Float16)Wz[k * 64 + 16 * t + col]; // Wz row-major (128,64)
        }
    } else {
        int u = tile - 16;
        #pragma unroll
        for (int j = 0; j < 16; ++j) {
            int k = j + (j & 8) + 8 * h;                   // K = 32 single chunk
            dst[j] = (_Float16)Wr[k * 192 + 16 * u + col]; // Wr row-major (32,192)
        }
    }
}

// ---------------------------------------------------------------------------
// Fused main kernel: one wave32 per 16-edge tile.
//   A-matrix (16x32 f16) register layout per ISA: lane half h, element j:
//     k = j + (j & 8) + 8*h   (rows = lane & 15)
//   C/D (16x16 f32): lane l holds col n = l&15, rows m = v + 8*(l>>4), v=0..7
// ---------------------------------------------------------------------------
__global__ __launch_bounds__(WAVES_PER_BLOCK * 32)
void edge_embedding_kernel(const float* __restrict__ vectors,
                           const int*   __restrict__ nl0,
                           const int*   __restrict__ nl1,
                           const int*   __restrict__ Z,
                           const float* __restrict__ z_table,
                           const float* __restrict__ br,
                           const float* __restrict__ beta,
                           const float* __restrict__ mu,
                           const _Float16* __restrict__ bw,
                           float* __restrict__ out, int E) {
    const int lane = threadIdx.x & 31;
    const int wave = threadIdx.x >> 5;
    const int tile = blockIdx.x * WAVES_PER_BLOCK + wave;
    const int base = tile * EDGES_PER_WAVE;
    if (base >= E) return;

    const int h   = lane >> 4;
    const int m16 = lane & 15;
    const int e   = base + m16;        // lanes 0-15 and 16-31 duplicate edges

    // ---- per-edge geometry (duplicated across lane halves) ----
    float vx = vectors[3 * e + 0];
    float vy = vectors[3 * e + 1];
    float vz = vectors[3 * e + 2];
    float r    = sqrtf(vx * vx + vy * vy + vz * vz);
    float rinv = 1.0f / r;
    float xh = vx * rinv, yh = vy * rinv, zh = vz * rinv;
    float er  = __expf(-r);
    float env = (r < CUTOFF_F) ? 0.5f * (__cosf(r * (PI_F / CUTOFF_F)) + 1.0f)
                               : 0.0f;

    // ---- rbf A-tile (16 edges x 32 K) in f16 ----
    v16h a_rbf;
    #pragma unroll
    for (int j = 0; j < 16; ++j) {
        int k   = j + (j & 8) + 8 * h;
        float d = er - mu[k];
        a_rbf[j] = (_Float16)__expf(-beta[k] * d * d);
    }

    const v8f zacc = {0.f, 0.f, 0.f, 0.f, 0.f, 0.f, 0.f, 0.f};

    // ---- h_r = rbf @ Wr : 12 n-tiles, single K=32 chunk ----
    v8f hr[12];
    #pragma unroll
    for (int u = 0; u < 12; ++u) {
        v16h b = *(const v16h*)(bw + (size_t)(16 + u) * 512 + (size_t)lane * 16);
        hr[u] = __builtin_amdgcn_wmma_f32_16x16x32_f16(
            false, a_rbf, false, b, (short)0, zacc, false, false);
    }

    // ---- h_z = h_edge @ Wz : 4 K-chunks x 4 n-tiles ----
    const float* row0 = z_table + (size_t)Z[nl0[e]] * 64;
    const float* row1 = z_table + (size_t)Z[nl1[e]] * 64;
    v8f hz[4] = {zacc, zacc, zacc, zacc};
    #pragma unroll
    for (int kc = 0; kc < 4; ++kc) {
        const float* row = (kc < 2) ? row0 : row1;
        int kbase = 32 * (kc & 1);
        v16h a;
        #pragma unroll
        for (int j = 0; j < 16; ++j) {
            int k = kbase + j + (j & 8) + 8 * h;
            a[j] = (_Float16)row[k];
        }
        #pragma unroll
        for (int t = 0; t < 4; ++t) {
            v16h b = *(const v16h*)(bw + (size_t)(kc * 4 + t) * 512 +
                                    (size_t)lane * 16);
            hz[t] = __builtin_amdgcn_wmma_f32_16x16x32_f16(
                false, a, false, b, (short)0, hz[t], false, false);
        }
    }

    // ---- redistribute per-edge scalars to C/D row layout via shuffles ----
    float xm[8], ym[8], zm[8], envm[8];
    #pragma unroll
    for (int v = 0; v < 8; ++v) {
        int src = v + 8 * h;           // lanes 0-15 hold edge base+src
        xm[v]   = __shfl(xh,  src, 32);
        ym[v]   = __shfl(yh,  src, 32);
        zm[v]   = __shfl(zh,  src, 32);
        envm[v] = __shfl(env, src, 32);
    }

    // ---- per-lane bias values (column = 16u + m16) ----
    float brv[12];
    #pragma unroll
    for (int u = 0; u < 12; ++u) brv[u] = br[16 * u + m16];

    // ---- expand + store: out_I / out_A / out_S each (E,64,3,3) ----
    const size_t sect = (size_t)E * 576;   // 64*9 floats per edge per tensor
    const float third = 1.0f / 3.0f;
    #pragma unroll
    for (int t = 0; t < 4; ++t) {
        #pragma unroll
        for (int v = 0; v < 8; ++v) {
            float en  = envm[v];
            float hzv = hz[t][v];
            float cI  = (hr[t    ][v] + brv[t    ]) * en * hzv;
            float cA  = (hr[t + 4][v] + brv[t + 4]) * en * hzv;
            float cS  = (hr[t + 8][v] + brv[t + 8]) * en * hzv;
            int   edge = base + v + 8 * h;
            int   ch   = 16 * t + m16;
            size_t o   = ((size_t)edge * 64 + (size_t)ch) * 9;
            float* pI = out + o;
            float* pA = out + sect + o;
            float* pS = out + 2 * sect + o;
            float x = xm[v], y = ym[v], z = zm[v];
            // out_I = cI * eye
            pI[0] = cI;  pI[1] = 0.f; pI[2] = 0.f;
            pI[3] = 0.f; pI[4] = cI;  pI[5] = 0.f;
            pI[6] = 0.f; pI[7] = 0.f; pI[8] = cI;
            // out_A = cA * [[0,-z,y],[z,0,-x],[-y,x,0]]
            pA[0] = 0.f;     pA[1] = -z * cA; pA[2] =  y * cA;
            pA[3] =  z * cA; pA[4] = 0.f;     pA[5] = -x * cA;
            pA[6] = -y * cA; pA[7] =  x * cA; pA[8] = 0.f;
            // out_S = cS * (r_hat r_hat^T - I/3)
            pS[0] = (x * x - third) * cS; pS[1] = x * y * cS;           pS[2] = x * z * cS;
            pS[3] = x * y * cS;           pS[4] = (y * y - third) * cS; pS[5] = y * z * cS;
            pS[6] = x * z * cS;           pS[7] = y * z * cS;           pS[8] = (z * z - third) * cS;
        }
    }
}

extern "C" void kernel_launch(void* const* d_in, const int* in_sizes, int n_in,
                              void* d_out, int out_size, void* d_ws, size_t ws_size,
                              hipStream_t stream) {
    const float* vectors = (const float*)d_in[0];   // (E,3)
    const int*   Z       = (const int*)  d_in[1];   // (N,)
    const int*   nl      = (const int*)  d_in[2];   // (2,E)
    const float* z_table = (const float*)d_in[3];   // (119,64)
    const float* Wz      = (const float*)d_in[4];   // (128,64)
    const float* Wr      = (const float*)d_in[5];   // (32,192)
    const float* br      = (const float*)d_in[6];   // (192,)
    const float* beta    = (const float*)d_in[7];   // (32,)
    const float* mu      = (const float*)d_in[8];   // (32,)
    float* out = (float*)d_out;

    const int E = in_sizes[2] / 2;
    _Float16* bw = (_Float16*)d_ws;                 // 28 tiles * 512 halfs = 28 KB

    // Repack weights into f16 WMMA B-tile register images (28 tiles).
    pack_weights_kernel<<<28, 32, 0, stream>>>(Wz, Wr, bw);

    // One wave32 per 16-edge tile; 8 waves per block.
    const int tiles  = (E + EDGES_PER_WAVE - 1) / EDGES_PER_WAVE;
    const int blocks = (tiles + WAVES_PER_BLOCK - 1) / WAVES_PER_BLOCK;
    edge_embedding_kernel<<<blocks, WAVES_PER_BLOCK * 32, 0, stream>>>(
        vectors, nl, nl + E, Z, z_table, br, beta, mu, bw, out, E);
}
